// SDPA_463856468443
// MI455X (gfx1250) — compile-verified, matrix operand order checked by reading, not run on hardware
//
#include <hip/hip_runtime.h>
#include <math.h>

typedef __attribute__((ext_vector_type(16))) _Float16 v16h;
typedef __attribute__((ext_vector_type(8)))  _Float16 v8h;
typedef __attribute__((ext_vector_type(4)))  _Float16 v4h;
typedef __attribute__((ext_vector_type(8)))  float    v8f;

#define B_SZ 4
#define H_SZ 16
#define S_LEN 2048
#define D_HEAD 128
#define QT 128           // query rows per block (halves K/V re-read traffic)
#define KTILE 64         // key cols per iteration
#define NTILES (S_LEN / KTILE)
#define KS  136          // K tile row stride (halfs), layout [key][d]   (136 % 8 == 0)
#define VTS 72           // V^T tile row stride (halfs), layout [d][key] (72 % 8 == 0)
#define SCS 68           // score row stride (floats)
#define PS  72           // pbuf row stride (halfs)                      (72 % 8 == 0)
#define SCALE 0.08838834764831845f   // 1/sqrt(128)

static __device__ __forceinline__ v16h frag_cat(v8h lo, v8h hi) {
    return __builtin_shufflevector(lo, hi, 0, 1, 2, 3, 4, 5, 6, 7,
                                           8, 9, 10, 11, 12, 13, 14, 15);
}

__global__ __launch_bounds__(256) void sdpa_flash_wmma(
    const float* __restrict__ q,
    const float* __restrict__ k,
    const float* __restrict__ v,
    const int*   __restrict__ mask,
    float*       __restrict__ out)
{
    __shared__ alignas(16) _Float16 ktile[KTILE * KS];    // K tile, natural [key][d]
    __shared__ alignas(16) _Float16 vtr[D_HEAD * VTS];    // V tile transposed [d][key]
    __shared__ alignas(16) float    sc[QT * SCS];         // raw scores (f32)
    __shared__ alignas(16) _Float16 pbuf[QT * PS];        // softmax probs (f16)
    __shared__ float m_row[QT], l_row[QT], c_row[QT];
    __shared__ float pm[QT * 2], pl[QT * 2];              // per-half partials

    const int tid  = threadIdx.x;
    const int lane = tid & 31;          // wave32
    const int wave = tid >> 5;          // 8 waves; wave == M subtile (16 rows each)

    const int bid   = blockIdx.x;
    const int qtile = bid & (S_LEN / QT - 1);   // 16 query tiles
    const int bh    = bid >> 4;                 // (b*H + h), 0..63
    const int q0    = qtile * QT;

    const float* qb = q + (size_t)bh * S_LEN * D_HEAD;
    const float* kb = k + (size_t)bh * S_LEN * D_HEAD;
    const float* vb = v + (size_t)bh * S_LEN * D_HEAD;
    float*       ob = out + (size_t)bh * S_LEN * D_HEAD;

    const int lan16 = lane & 15;
    const int khi   = (lane & 16) ? 8 : 0;   // half-lane K (A/B) / M (C/D) offset

    // ---- Q A-fragments: this wave's 16 rows x 128 d as 4x v16h (b128 global loads).
    const int arow = q0 + wave * 16 + lan16;
    v16h qfrag[4];
#pragma unroll
    for (int f = 0; f < 4; ++f) {
        const float* qp = &qb[(size_t)arow * D_HEAD + f * 32 + khi];
        const float4 a0 = *(const float4*)(qp);
        const float4 a1 = *(const float4*)(qp + 4);
        const float4 a2 = *(const float4*)(qp + 16);
        const float4 a3 = *(const float4*)(qp + 20);
        v16h qa;
        qa[0]  = (_Float16)a0.x; qa[1]  = (_Float16)a0.y;
        qa[2]  = (_Float16)a0.z; qa[3]  = (_Float16)a0.w;
        qa[4]  = (_Float16)a1.x; qa[5]  = (_Float16)a1.y;
        qa[6]  = (_Float16)a1.z; qa[7]  = (_Float16)a1.w;
        qa[8]  = (_Float16)a2.x; qa[9]  = (_Float16)a2.y;
        qa[10] = (_Float16)a2.z; qa[11] = (_Float16)a2.w;
        qa[12] = (_Float16)a3.x; qa[13] = (_Float16)a3.y;
        qa[14] = (_Float16)a3.z; qa[15] = (_Float16)a3.w;
        qfrag[f] = qa;
    }

    // ---- Output accumulators: full 128-wide row block -> 8 subtiles of 16x16.
    v8f oacc[8];
#pragma unroll
    for (int s = 0; s < 8; ++s)
#pragma unroll
        for (int r = 0; r < 8; ++r) oacc[s][r] = 0.0f;

    for (int i = tid; i < QT; i += 256) {
        m_row[i] = -INFINITY; l_row[i] = 0.0f; c_row[i] = 1.0f;
    }
    __syncthreads();

    const int sr = tid >> 1;   // softmax row (0..127)
    const int sq = tid & 1;    // half (32 cols) within the row

    for (int t = 0; t < NTILES; ++t) {
        const int k0 = t * KTILE;

        // ---- Stage tiles: K natural (b128 read -> b64 LDS store),
        //      V transposed (b128 read -> 4x b16 scattered stores).
        for (int i = tid; i < (KTILE * D_HEAD) / 4; i += 256) {
            const int key = i >> 5;
            const int dc  = (i & 31) << 2;
            const float4 kf = *(const float4*)&kb[(size_t)(k0 + key) * D_HEAD + dc];
            const float4 vf = *(const float4*)&vb[(size_t)(k0 + key) * D_HEAD + dc];
            v4h kh;
            kh[0] = (_Float16)kf.x; kh[1] = (_Float16)kf.y;
            kh[2] = (_Float16)kf.z; kh[3] = (_Float16)kf.w;
            *(v4h*)&ktile[key * KS + dc] = kh;
            vtr[(dc + 0) * VTS + key] = (_Float16)vf.x;
            vtr[(dc + 1) * VTS + key] = (_Float16)vf.y;
            vtr[(dc + 2) * VTS + key] = (_Float16)vf.z;
            vtr[(dc + 3) * VTS + key] = (_Float16)vf.w;
        }
        if (t + 1 < NTILES && tid < KTILE) {
            __builtin_prefetch(&kb[(size_t)(k0 + KTILE + tid) * D_HEAD], 0, 1);
            __builtin_prefetch(&vb[(size_t)(k0 + KTILE + tid) * D_HEAD], 0, 1);
        }
        __syncthreads();

        // ---- Scores: each wave computes its 16 rows x 64 keys = 4 subtiles.
        const int mb = wave * 16 + khi;
#pragma unroll
        for (int nt = 0; nt < 4; ++nt) {
            const int n = nt * 16 + lan16;
            v8f c;
#pragma unroll
            for (int r = 0; r < 8; ++r) c[r] = 0.0f;
#pragma unroll
            for (int f = 0; f < 4; ++f) {
                const _Float16* bp = &ktile[n * KS + f * 32 + khi];
                const v16h bfr = frag_cat(*(const v8h*)bp, *(const v8h*)(bp + 16));
                c = __builtin_amdgcn_wmma_f32_16x16x32_f16(
                        false, qfrag[f], false, bfr, (short)0, c, false, false);
            }
#pragma unroll
            for (int r = 0; r < 8; ++r)
                sc[(mb + r) * SCS + n] = c[r];
        }
        __syncthreads();

        // ---- Online softmax, 2 threads per row, fully vectorized.
        // Pass 1: scale + mask + partial max (int4 mask, float4 scores).
        {
            const int4* mg = (const int4*)(mask + (size_t)(q0 + sr) * S_LEN + k0 + sq * 32);
            float mx = -INFINITY;
#pragma unroll
            for (int j4 = 0; j4 < 8; ++j4) {
                float4* sp = (float4*)&sc[sr * SCS + sq * 32 + j4 * 4];
                float4 x4 = *sp;
                const int4 m4 = mg[j4];
                x4.x = (m4.x == 0) ? -INFINITY : x4.x * SCALE;
                x4.y = (m4.y == 0) ? -INFINITY : x4.y * SCALE;
                x4.z = (m4.z == 0) ? -INFINITY : x4.z * SCALE;
                x4.w = (m4.w == 0) ? -INFINITY : x4.w * SCALE;
                *sp = x4;
                mx = fmaxf(mx, fmaxf(fmaxf(x4.x, x4.y), fmaxf(x4.z, x4.w)));
            }
            pm[sr * 2 + sq] = mx;
        }
        __syncthreads();
        if (sq == 0) {
            const float mo = m_row[sr];
            float mn = fmaxf(fmaxf(pm[sr * 2], pm[sr * 2 + 1]), mo);
            c_row[sr] = (mn > -INFINITY) ? __expf(mo - mn) : 1.0f;
            m_row[sr] = mn;
        }
        __syncthreads();
        // Pass 2: exponentiate, emit f16 probs (v8h stores), partial sums.
        {
            const float mn = m_row[sr];
            float ls = 0.0f;
#pragma unroll
            for (int c8 = 0; c8 < 4; ++c8) {
                v8h ph;
#pragma unroll
                for (int j = 0; j < 8; ++j) {
                    const float x = sc[sr * SCS + sq * 32 + c8 * 8 + j];
                    const float p = (x > -INFINITY) ? __expf(x - mn) : 0.0f;
                    ph[j] = (_Float16)p; ls += p;
                }
                *(v8h*)&pbuf[sr * PS + sq * 32 + c8 * 8] = ph;
            }
            pl[sr * 2 + sq] = ls;
        }
        __syncthreads();
        for (int i = tid; i < QT; i += 256)
            l_row[i] = l_row[i] * c_row[i] + (pl[i * 2] + pl[i * 2 + 1]);

        // ---- Rescale O accumulators by per-row correction factor.
#pragma unroll
        for (int r = 0; r < 8; ++r) {
            const float fct = c_row[mb + r];
#pragma unroll
            for (int s = 0; s < 8; ++s) oacc[s][r] *= fct;
        }

        // ---- O += P(16x64) * V(64x128): 8 output subtiles; A-fragments reused.
        const int mrowA = wave * 16 + lan16;
#pragma unroll
        for (int f = 0; f < 2; ++f) {
            const _Float16* pa = &pbuf[mrowA * PS + f * 32 + khi];
            const v16h a = frag_cat(*(const v8h*)pa, *(const v8h*)(pa + 16));
#pragma unroll
            for (int s = 0; s < 8; ++s) {
                const int ncol = s * 16 + lan16;
                const _Float16* pv = &vtr[ncol * VTS + f * 32 + khi];
                const v16h bfr = frag_cat(*(const v8h*)pv, *(const v8h*)(pv + 16));
                oacc[s] = __builtin_amdgcn_wmma_f32_16x16x32_f16(
                              false, a, false, bfr, (short)0, oacc[s], false, false);
            }
        }
        __syncthreads();   // protect LDS tiles + l_row before next iteration
    }

    // ---- Epilogue: normalize by row sums, store fp32.
    const int mb = wave * 16 + khi;
#pragma unroll
    for (int r = 0; r < 8; ++r) {
        const float ls  = l_row[mb + r];
        const float inv = (ls > 0.0f) ? (1.0f / ls) : 0.0f;
        const int   row = q0 + mb + r;
#pragma unroll
        for (int s = 0; s < 8; ++s) {
            const int col = s * 16 + lan16;
            ob[(size_t)row * D_HEAD + col] = oacc[s][r] * inv;
        }
    }
}

extern "C" void kernel_launch(void* const* d_in, const int* in_sizes, int n_in,
                              void* d_out, int out_size, void* d_ws, size_t ws_size,
                              hipStream_t stream) {
    const float* q    = (const float*)d_in[0];
    const float* k    = (const float*)d_in[1];
    const float* v    = (const float*)d_in[2];
    const int*   mask = (const int*)d_in[3];
    float*       out  = (float*)d_out;

    const int blocks = B_SZ * H_SZ * (S_LEN / QT);   // 4*16*16 = 1024
    sdpa_flash_wmma<<<dim3(blocks), dim3(256), 0, stream>>>(q, k, v, mask, out);
}
